// BMEDLSTM_60172491817048
// MI455X (gfx1250) — compile-verified, matrix operand order checked by reading, not run on hardware
//
#include <hip/hip_runtime.h>
#include <hip/hip_bf16.h>
#include <stdint.h>

// ---------------------------------------------------------------------------
// Persistent 2-layer LSTM for MI455X (gfx1250, wave32, WMMA).
//  grid = 16 batch-tiles x 8 column-groups = 128 blocks, 128 threads (4 waves)
//  per wave: 16 hidden columns -> 4 gate N-tiles; WMMA f32_16x16x32_bf16
//  weights converted once to bf16 (L2-resident), h0/h1 exchanged via
//  double-buffered global bf16 buffers + 8-block atomic phase barriers.
//  x_t tiles staged into LDS via the Tensor Data Mover (double buffered).
// ---------------------------------------------------------------------------

typedef __bf16 bf16_t;
typedef __attribute__((ext_vector_type(16))) __bf16 v16bf;
typedef __attribute__((ext_vector_type(8)))  __bf16 v8bf;
typedef __attribute__((ext_vector_type(8)))  float  v8f;
typedef __attribute__((ext_vector_type(4)))  unsigned int u32x4;
typedef __attribute__((ext_vector_type(8)))  int  i32x8;
typedef __attribute__((ext_vector_type(4)))  int  i32x4;

#define HID   512
#define XS    128
#define SS    64
#define IN0   (XS + SS)          // 192
#define BATCH 256
#define TROWS 16
#define NGRP  8                  // column-group blocks per batch tile
#define NTILES (BATCH / TROWS)   // 16
#define NBLOCKS (NTILES * NGRP)  // 128
#define NTHR  128                // 4 waves of 32

#if defined(__gfx1250__) && __has_builtin(__builtin_amdgcn_tensor_load_to_lds) && __has_builtin(__builtin_amdgcn_s_wait_tensorcnt)
#define USE_TDM 1
#else
#define USE_TDM 0
#endif

// ---------------- fragment helpers (wave32 WMMA layouts) -------------------
// A (16x32 bf16): lane l<16 -> row M=l, K = {0..7, 16..23}; lane l>=16 -> row
// M=l-16, K = {8..15, 24..31}.  B (32x16) mirrors with lane = column N.
// Both reduce to two contiguous 16B loads per lane from a row-major matrix.
__device__ __forceinline__ v16bf load_frag16(const bf16_t* p) {
  v8bf lo = *(const v8bf*)p;
  v8bf hi = *(const v8bf*)(p + 16);
  v16bf r;
#pragma unroll
  for (int i = 0; i < 8; ++i) { r[i] = lo[i]; r[i + 8] = hi[i]; }
  return r;
}

__device__ __forceinline__ v8f wmma_bf16(v16bf a, v16bf b, v8f c) {
  return __builtin_amdgcn_wmma_f32_16x16x32_bf16(false, a, false, b,
                                                 (short)0, c, false, false);
}

__device__ __forceinline__ float fast_sig(float x) {
  return __builtin_amdgcn_rcpf(1.0f + __builtin_amdgcn_exp2f(-1.44269504088896f * x));
}
__device__ __forceinline__ float fast_tanh(float x) {
  return 2.0f * fast_sig(2.0f * x) - 1.0f;
}

// ---------------- TDM 2D tile load (f32 elements) --------------------------
#if USE_TDM
__device__ __forceinline__ void tdm_load_2d_f32(unsigned lds_off, const void* gptr,
                                                unsigned width_elts, unsigned rows,
                                                unsigned long long row_stride_elts) {
  unsigned long long ga = (unsigned long long)(uintptr_t)gptr;
  u32x4 g0;
  g0[0] = 1u;                                            // count = 1 valid D#
  g0[1] = lds_off;                                       // LDS byte address
  g0[2] = (unsigned)(ga & 0xFFFFFFFFu);                  // global_addr lo
  g0[3] = (unsigned)((ga >> 32) & 0x01FFFFFFu) | (2u << 30); // addr hi | type=2
  unsigned w0 = (2u << 16);                              // data_size = 4B
  unsigned w1 = (width_elts & 0xFFFFu) << 16;            // tensor_dim0 lo16
  unsigned w2 = ((width_elts >> 16) & 0xFFFFu) | ((rows & 0xFFFFu) << 16);
  unsigned w3 = ((rows >> 16) & 0xFFFFu) | ((width_elts & 0xFFFFu) << 16); // tile_dim0
  unsigned w4 = rows & 0xFFFFu;                          // tile_dim1 (tile_dim2=0)
  unsigned w5 = (unsigned)(row_stride_elts & 0xFFFFFFFFull);
  unsigned w6 = (unsigned)((row_stride_elts >> 32) & 0xFFFFull);
  i32x8 g1;
  g1[0] = (int)w0; g1[1] = (int)w1; g1[2] = (int)w2; g1[3] = (int)w3;
  g1[4] = (int)w4; g1[5] = (int)w5; g1[6] = (int)w6; g1[7] = 0;
  i32x4 z4 = {0, 0, 0, 0};
  i32x8 z8 = {0, 0, 0, 0, 0, 0, 0, 0};
  // 6-arg (clang-23 / therock-10.0) form: groups 2,3 + extra operand group.
  __builtin_amdgcn_tensor_load_to_lds(g0, g1, z4, z4, z8, 0);
}
#endif

// ---------------- 8-block phase barrier ------------------------------------
__device__ __forceinline__ void group_barrier(int* ctr, int target) {
  __threadfence();
  __syncthreads();
  if (threadIdx.x == 0) {
    __hip_atomic_fetch_add(ctr, 1, __ATOMIC_RELEASE, __HIP_MEMORY_SCOPE_AGENT);
    while (__hip_atomic_load(ctr, __ATOMIC_ACQUIRE, __HIP_MEMORY_SCOPE_AGENT) < target) {
      __builtin_amdgcn_s_sleep(1);
    }
  }
  __syncthreads();
}

// ---------------- prep: fp32 -> bf16 weights, fused bias, zero state -------
__global__ void lstm_prep(const float* __restrict__ Wi0f, const float* __restrict__ Wh0f,
                          const float* __restrict__ Wi1f, const float* __restrict__ Wh1f,
                          const float* __restrict__ Wfcf,
                          const float* __restrict__ bi0, const float* __restrict__ bh0,
                          const float* __restrict__ bi1, const float* __restrict__ bh1,
                          const float* __restrict__ bfcf,
                          bf16_t* __restrict__ Wi0, bf16_t* __restrict__ Wh0,
                          bf16_t* __restrict__ Wi1, bf16_t* __restrict__ Wh1,
                          bf16_t* __restrict__ Wfc,
                          float* __restrict__ bias0, float* __restrict__ bias1,
                          float* __restrict__ bfc,
                          bf16_t* __restrict__ h0g, bf16_t* __restrict__ h1g,
                          int* __restrict__ ctrs) {
  const int tid = blockIdx.x * blockDim.x + threadIdx.x;
  const int np  = gridDim.x * blockDim.x;
  const int NWi0 = 4 * HID * IN0;        // 393216
  const int NWh  = 4 * HID * HID;        // 1048576
  const int NWfc = SS * HID;             // 32768
  const int NH   = 2 * BATCH * HID;      // 262144 (double buffered)
  for (int i = tid; i < NWi0; i += np) Wi0[i] = (bf16_t)Wi0f[i];
  for (int i = tid; i < NWh;  i += np) Wh0[i] = (bf16_t)Wh0f[i];
  for (int i = tid; i < NWh;  i += np) Wi1[i] = (bf16_t)Wi1f[i];
  for (int i = tid; i < NWh;  i += np) Wh1[i] = (bf16_t)Wh1f[i];
  for (int i = tid; i < NWfc; i += np) Wfc[i] = (bf16_t)Wfcf[i];
  for (int i = tid; i < 4 * HID; i += np) bias0[i] = bi0[i] + bh0[i];
  for (int i = tid; i < 4 * HID; i += np) bias1[i] = bi1[i] + bh1[i];
  for (int i = tid; i < SS; i += np) bfc[i] = bfcf[i];
  for (int i = tid; i < NH; i += np) { h0g[i] = (bf16_t)0.0f; h1g[i] = (bf16_t)0.0f; }
  if (tid < 64) ctrs[tid] = 0;
}

// ---------------- main persistent kernel -----------------------------------
__global__ __launch_bounds__(NTHR, 1)
void lstm_persistent(const float* __restrict__ x, const float* __restrict__ s0,
                     const bf16_t* __restrict__ Wi0, const bf16_t* __restrict__ Wh0,
                     const bf16_t* __restrict__ Wi1, const bf16_t* __restrict__ Wh1,
                     const bf16_t* __restrict__ Wfc,
                     const float* __restrict__ bias0, const float* __restrict__ bias1,
                     const float* __restrict__ bfc,
                     bf16_t* __restrict__ h0g, bf16_t* __restrict__ h1g,
                     int* __restrict__ ctrs, float* __restrict__ out,
                     const int* __restrict__ seqp, int Tdim, int Tout) {
  __shared__ __align__(16) bf16_t inp_lds[TROWS * IN0];   // [16][192] bf16
  __shared__ float s_lds[TROWS * SS];                     // [16][64] f32 running state
#if USE_TDM
  __shared__ __align__(16) float xstage[2][TROWS * XS];   // double-buffered x tile
#endif

  const int tile = blockIdx.x >> 3;       // 0..15  (batch rows tile*16..)
  const int cg   = blockIdx.x & 7;        // 0..7   (hidden cols cg*64..)
  const int wav  = threadIdx.x >> 5;      // 0..3
  const int lane = threadIdx.x & 31;
  const int idx  = lane & 15;             // M (A/acc) or N (B) index
  const int kb   = (lane >> 4) << 3;      // K octet select (0 or 8)
  const int mh   = (lane >> 4) << 3;      // acc M-half offset
  const int hc   = cg * 64 + wav * 16;    // hidden column base of this wave
  const int row0 = tile * TROWS;
  const int T    = seqp[0];

  // initial residual state s0 -> LDS (f32, accumulated in f32 all T steps)
  for (int i = threadIdx.x; i < TROWS * SS; i += NTHR)
    s_lds[i] = s0[(row0 + i / SS) * SS + (i % SS)];

  // persistent per-lane element offsets (gate rows g*H + hc + idx) and biases
  size_t oWi0[4], oWh0[4], oWi1[4], oWh1[4];
  float bs0[4], bs1[4];
#pragma unroll
  for (int g = 0; g < 4; ++g) {
    int r = g * HID + hc + idx;
    oWi0[g] = (size_t)r * IN0 + kb;
    oWh0[g] = (size_t)r * HID + kb;
    oWi1[g] = (size_t)r * HID + kb;
    oWh1[g] = (size_t)r * HID + kb;
    bs0[g] = bias0[r];
    bs1[g] = bias1[r];
  }
  const size_t oWfc = (size_t)(wav * 16 + idx) * HID + kb;
  const float  bsfc = bfc[wav * 16 + idx];

  v8f c0 = {}; v8f c1 = {};               // cell states live in registers
  int* ctr = ctrs + tile;
  int bar = 0;

#if USE_TDM
  if (threadIdx.x == 0) {
    tdm_load_2d_f32((unsigned)(uintptr_t)&xstage[0][0],
                    x + ((size_t)row0 * Tdim + 0) * XS,
                    XS, TROWS, (unsigned long long)Tdim * XS);
  }
#endif

  for (int t = 0; t < T; ++t) {
    const int cur = t & 1, nxt = cur ^ 1; // h(t) -> buf cur, h(t-1) -> buf nxt

    // Opaque zero offset, re-materialized every iteration: makes the weight
    // fragment addresses loop-variant so LICM cannot hoist the ~232 fragment
    // loads out of the time loop (-> scratch spills), while keeping pointer
    // provenance intact so loads stay GLOBAL (not FLAT).
    unsigned zoff = 0;
    asm volatile("" : "+v"(zoff));
    const bf16_t* wWi0[4]; const bf16_t* wWh0[4];
    const bf16_t* wWi1[4]; const bf16_t* wWh1[4];
#pragma unroll
    for (int g = 0; g < 4; ++g) {
      wWi0[g] = Wi0 + (oWi0[g] + zoff);
      wWh0[g] = Wh0 + (oWh0[g] + zoff);
      wWi1[g] = Wi1 + (oWi1[g] + zoff);
      wWh1[g] = Wh1 + (oWh1[g] + zoff);
    }
    const bf16_t* wWfc = Wfc + (oWfc + zoff);

    // ---- stage inputs: inp = [bf16(x_t) | bf16(s)] -------------------------
#if USE_TDM
    if (threadIdx.x == 0) __builtin_amdgcn_s_wait_tensorcnt(0);
#endif
    __syncthreads();
    for (int i = threadIdx.x; i < TROWS * XS; i += NTHR) {
      int m = i / XS, k = i % XS;
#if USE_TDM
      float v = xstage[cur][i];
#else
      float v = x[((size_t)(row0 + m) * Tdim + t) * XS + k];
#endif
      inp_lds[m * IN0 + k] = (bf16_t)v;
    }
    for (int i = threadIdx.x; i < TROWS * SS; i += NTHR)
      inp_lds[(i / SS) * IN0 + XS + (i % SS)] = (bf16_t)s_lds[i];
    __syncthreads();
#if USE_TDM
    if (t + 1 < T && threadIdx.x == 0) {
      tdm_load_2d_f32((unsigned)(uintptr_t)&xstage[nxt][0],
                      x + ((size_t)row0 * Tdim + (t + 1)) * XS,
                      XS, TROWS, (unsigned long long)Tdim * XS);
    }
#endif

    // ---- layer 0 gates: [16 x 4x16] = inp(192) + h0(t-1)(512) -------------
    v8f acc0 = {}, acc1 = {}, acc2 = {}, acc3 = {};
    {
      const bf16_t* aI = inp_lds + idx * IN0 + kb;
#pragma unroll
      for (int kt = 0; kt < IN0; kt += 32) {
        v16bf af = load_frag16(aI + kt);
        acc0 = wmma_bf16(af, load_frag16(wWi0[0] + kt), acc0);
        acc1 = wmma_bf16(af, load_frag16(wWi0[1] + kt), acc1);
        acc2 = wmma_bf16(af, load_frag16(wWi0[2] + kt), acc2);
        acc3 = wmma_bf16(af, load_frag16(wWi0[3] + kt), acc3);
      }
      const bf16_t* aH = h0g + ((size_t)(nxt * BATCH + row0 + idx)) * HID + kb;
#pragma unroll
      for (int kt = 0; kt < HID; kt += 32) {
        v16bf af = load_frag16(aH + kt);
        acc0 = wmma_bf16(af, load_frag16(wWh0[0] + kt), acc0);
        acc1 = wmma_bf16(af, load_frag16(wWh0[1] + kt), acc1);
        acc2 = wmma_bf16(af, load_frag16(wWh0[2] + kt), acc2);
        acc3 = wmma_bf16(af, load_frag16(wWh0[3] + kt), acc3);
      }
    }
    {
      bf16_t* h0w = h0g + ((size_t)(cur * BATCH + row0)) * HID + hc + idx;
#pragma unroll
      for (int v = 0; v < 8; ++v) {
        float cv = fast_sig(acc1[v] + bs0[1]) * c0[v] +
                   fast_sig(acc0[v] + bs0[0]) * fast_tanh(acc2[v] + bs0[2]);
        c0[v] = cv;
        float hv = fast_sig(acc3[v] + bs0[3]) * fast_tanh(cv);
        h0w[(size_t)(v + mh) * HID] = (bf16_t)hv;
      }
    }
    bar += NGRP; group_barrier(ctr, bar);   // h0(t) now complete for the tile

    // ---- layer 1 gates: h0(t)(512) + h1(t-1)(512) -------------------------
    acc0 = (v8f){}; acc1 = (v8f){}; acc2 = (v8f){}; acc3 = (v8f){};
    {
      const bf16_t* a0 = h0g + ((size_t)(cur * BATCH + row0 + idx)) * HID + kb;
#pragma unroll
      for (int kt = 0; kt < HID; kt += 32) {
        v16bf af = load_frag16(a0 + kt);
        acc0 = wmma_bf16(af, load_frag16(wWi1[0] + kt), acc0);
        acc1 = wmma_bf16(af, load_frag16(wWi1[1] + kt), acc1);
        acc2 = wmma_bf16(af, load_frag16(wWi1[2] + kt), acc2);
        acc3 = wmma_bf16(af, load_frag16(wWi1[3] + kt), acc3);
      }
      const bf16_t* a1 = h1g + ((size_t)(nxt * BATCH + row0 + idx)) * HID + kb;
#pragma unroll
      for (int kt = 0; kt < HID; kt += 32) {
        v16bf af = load_frag16(a1 + kt);
        acc0 = wmma_bf16(af, load_frag16(wWh1[0] + kt), acc0);
        acc1 = wmma_bf16(af, load_frag16(wWh1[1] + kt), acc1);
        acc2 = wmma_bf16(af, load_frag16(wWh1[2] + kt), acc2);
        acc3 = wmma_bf16(af, load_frag16(wWh1[3] + kt), acc3);
      }
    }
    {
      bf16_t* h1w = h1g + ((size_t)(cur * BATCH + row0)) * HID + hc + idx;
#pragma unroll
      for (int v = 0; v < 8; ++v) {
        float cv = fast_sig(acc1[v] + bs1[1]) * c1[v] +
                   fast_sig(acc0[v] + bs1[0]) * fast_tanh(acc2[v] + bs1[2]);
        c1[v] = cv;
        float hv = fast_sig(acc3[v] + bs1[3]) * fast_tanh(cv);
        h1w[(size_t)(v + mh) * HID] = (bf16_t)hv;
      }
    }
    bar += NGRP; group_barrier(ctr, bar);   // h1(t) now complete for the tile

    // ---- FC + residual: out = h1 @ Wfc.T + bfc ; s += out -----------------
    {
      v8f f = {};
      const bf16_t* a1 = h1g + ((size_t)(cur * BATCH + row0 + idx)) * HID + kb;
#pragma unroll
      for (int kt = 0; kt < HID; kt += 32)
        f = wmma_bf16(load_frag16(a1 + kt), load_frag16(wWfc + kt), f);
      const int n = wav * 16 + idx;
#pragma unroll
      for (int v = 0; v < 8; ++v) {
        float d = f[v] + bsfc;
        int m = v + mh;
        s_lds[m * SS + n] += d;                           // residual state (f32)
        if (cg == 0)
          out[((size_t)(row0 + m) * Tout + t) * SS + n] = d;
      }
    }
    __syncthreads();   // s_lds consistent before next step's input staging
  }
}

// ---------------------------------------------------------------------------
extern "C" void kernel_launch(void* const* d_in, const int* in_sizes, int n_in,
                              void* d_out, int out_size, void* d_ws, size_t ws_size,
                              hipStream_t stream) {
  const float* x    = (const float*)d_in[0];
  const float* s0   = (const float*)d_in[1];
  const float* Wi0f = (const float*)d_in[2];
  const float* Wh0f = (const float*)d_in[3];
  const float* bi0  = (const float*)d_in[4];
  const float* bh0  = (const float*)d_in[5];
  const float* Wi1f = (const float*)d_in[6];
  const float* Wh1f = (const float*)d_in[7];
  const float* bi1  = (const float*)d_in[8];
  const float* bh1  = (const float*)d_in[9];
  const float* Wfcf = (const float*)d_in[10];
  const float* bfcf = (const float*)d_in[11];
  const int*   seqp = (const int*)d_in[12];
  float* out = (float*)d_out;

  const int Tdim = in_sizes[0] / (BATCH * XS);   // time extent of x
  const int Tout = out_size   / (BATCH * SS);    // time extent of out

  char* ws = (char*)d_ws;
  size_t off = 0;
  auto carve = [&](size_t bytes) -> char* {
    char* p = ws + off;
    off = (off + bytes + 255) & ~(size_t)255;
    return p;
  };
  bf16_t* Wi0 = (bf16_t*)carve(sizeof(bf16_t) * 4 * HID * IN0);
  bf16_t* Wh0 = (bf16_t*)carve(sizeof(bf16_t) * 4 * HID * HID);
  bf16_t* Wi1 = (bf16_t*)carve(sizeof(bf16_t) * 4 * HID * HID);
  bf16_t* Wh1 = (bf16_t*)carve(sizeof(bf16_t) * 4 * HID * HID);
  bf16_t* Wfc = (bf16_t*)carve(sizeof(bf16_t) * SS * HID);
  float* bias0 = (float*)carve(sizeof(float) * 4 * HID);
  float* bias1 = (float*)carve(sizeof(float) * 4 * HID);
  float* bfc   = (float*)carve(sizeof(float) * SS);
  bf16_t* h0g  = (bf16_t*)carve(sizeof(bf16_t) * 2 * BATCH * HID);
  bf16_t* h1g  = (bf16_t*)carve(sizeof(bf16_t) * 2 * BATCH * HID);
  int* ctrs    = (int*)carve(sizeof(int) * 64);

  lstm_prep<<<512, 256, 0, stream>>>(Wi0f, Wh0f, Wi1f, Wh1f, Wfcf,
                                     bi0, bh0, bi1, bh1, bfcf,
                                     Wi0, Wh0, Wi1, Wh1, Wfc,
                                     bias0, bias1, bfc, h0g, h1g, ctrs);

  lstm_persistent<<<NBLOCKS, NTHR, 0, stream>>>(x, s0, Wi0, Wh0, Wi1, Wh1, Wfc,
                                                bias0, bias1, bfc, h0g, h1g,
                                                ctrs, out, seqp, Tdim, Tout);
}
